// NoisyRouter_88493506167190
// MI455X (gfx1250) — matrix operand order repaired and checked by reading, not compiled, not run on hardware
//
#include <hip/hip_runtime.h>
#include <hip/hip_bf16.h>

typedef __attribute__((ext_vector_type(16))) __bf16 v16bf;
typedef __attribute__((ext_vector_type(8)))  float  v8f;

#define EXPERTS 16

__device__ __forceinline__ v16bf pack16(const float4& a, const float4& b,
                                        const float4& c, const float4& d) {
  v16bf r;
  r[0]  = (__bf16)a.x; r[1]  = (__bf16)a.y; r[2]  = (__bf16)a.z; r[3]  = (__bf16)a.w;
  r[4]  = (__bf16)b.x; r[5]  = (__bf16)b.y; r[6]  = (__bf16)b.z; r[7]  = (__bf16)b.w;
  r[8]  = (__bf16)c.x; r[9]  = (__bf16)c.y; r[10] = (__bf16)c.z; r[11] = (__bf16)c.w;
  r[12] = (__bf16)d.x; r[13] = (__bf16)d.y; r[14] = (__bf16)d.z; r[15] = (__bf16)d.w;
  return r;
}

__global__ __launch_bounds__(256)
void noisy_router_kernel(const float* __restrict__ x,
                         const float* __restrict__ noise,
                         const float* __restrict__ Wfc,
                         const float* __restrict__ bfc,
                         const float* __restrict__ Wns,
                         const float* __restrict__ bns,
                         float* __restrict__ out,
                         int* __restrict__ ids,
                         int M, int D, int ntiles) {
  // per-wave staging of the two 16x16 f32 logit tiles (8 waves/block -> 16 KB)
  __shared__ float s_fc[8][16][16];
  __shared__ float s_ns[8][16][16];

  const int lane = threadIdx.x & 31;
  const int wave = threadIdx.x >> 5;
  const int l    = lane & 15;   // row (A) / expert column (B,C)
  const int h    = lane >> 4;   // half-wave selector for K / M split
  const int tile = blockIdx.x * 8 + wave;
  const bool active = tile < ntiles;  // wave-uniform

  if (active) {
    const int row0 = tile * 16;
    const float* xrow = x   + (size_t)(row0 + l) * D;  // A: this lane's row
    const float* wfr  = Wfc + (size_t)l * D;           // B: this lane's expert row
    const float* wnr  = Wns + (size_t)l * D;
    v8f cfc = {};
    v8f cns = {};
    for (int k0 = 0; k0 < D; k0 += 32) {
      // A 16x32 bf16: lanes 0-15 carry K = k0..k0+15, lanes 16-31 carry K = k0+16..k0+31
      const float4* pa0 = (const float4*)(xrow + k0 + h * 16);
      const float4* pa1 = pa0 + 2;
      float4 a0 = pa0[0], a1 = pa0[1];
      float4 a2 = pa1[0], a3 = pa1[1];
      // B 32x16 bf16: lanes 0-15 carry K = k0..k0+15, lanes 16-31 carry K = k0+16..k0+31
      const float4* pf = (const float4*)(wfr + k0 + h * 16);
      float4 f0 = pf[0], f1 = pf[1], f2 = pf[2], f3 = pf[3];
      const float4* pn = (const float4*)(wnr + k0 + h * 16);
      float4 n0 = pn[0], n1 = pn[1], n2 = pn[2], n3 = pn[3];
      if (k0 + 96 < D)
        __builtin_prefetch(xrow + k0 + 96, 0, 1);  // stream-ahead on the HBM-bound operand
      v16bf A  = pack16(a0, a1, a2, a3);
      v16bf Bf = pack16(f0, f1, f2, f3);
      v16bf Bn = pack16(n0, n1, n2, n3);
      cfc = __builtin_amdgcn_wmma_f32_16x16x32_bf16(false, A, false, Bf,
                                                    (short)0, cfc, false, false);
      cns = __builtin_amdgcn_wmma_f32_16x16x32_bf16(false, A, false, Bn,
                                                    (short)0, cns, false, false);
    }
    // C layout: VGPR r -> M = r + 8*h, N = lane&15
#pragma unroll
    for (int r = 0; r < 8; ++r) {
      s_fc[wave][r + 8 * h][l] = cfc[r];
      s_ns[wave][r + 8 * h][l] = cns[r];
    }
  }
  __syncthreads();

  if (active && h == 0) {
    const int row0 = tile * 16;
    const int g = row0 + l;          // global row owned by this lane
    float v0 = -3.0e38f, v1 = -3.0e38f;
    int   i0 = 0,        i1 = 0;
#pragma unroll
    for (int e = 0; e < EXPERTS; ++e) {
      float fc = s_fc[wave][l][e] + bfc[e];
      float ns = s_ns[wave][l][e] + bns[e];
      // numerically stable softplus
      float sp = fmaxf(ns, 0.f) + __logf(1.f + __expf(-fabsf(ns)));
      float nv = fc + noise[(size_t)g * EXPERTS + e] * sp;
      if (nv > v0)      { v1 = v0; i1 = i0; v0 = nv; i0 = e; }
      else if (nv > v1) { v1 = nv; i1 = e; }
    }
    // softmax over the two surviving logits (others are -inf -> 0)
    float d   = v1 - v0;
    float ed  = __expf(d);
    float inv = 1.f / (1.f + ed);
    float p0  = inv, p1 = ed * inv;
#pragma unroll
    for (int e = 0; e < EXPERTS; ++e) {
      float o = (e == i0) ? p0 : ((e == i1) ? p1 : 0.f);
      out[(size_t)g * EXPERTS + e] = o;
    }
    ids[(size_t)g * 2 + 0] = i0;
    ids[(size_t)g * 2 + 1] = i1;
  }
}

extern "C" void kernel_launch(void* const* d_in, const int* in_sizes, int n_in,
                              void* d_out, int out_size, void* d_ws, size_t ws_size,
                              hipStream_t stream) {
  const float* x     = (const float*)d_in[0];
  const float* noise = (const float*)d_in[1];
  const float* Wfc   = (const float*)d_in[2];
  const float* bfc   = (const float*)d_in[3];
  const float* Wns   = (const float*)d_in[4];
  const float* bns   = (const float*)d_in[5];

  const int M = in_sizes[1] / EXPERTS;   // noise is [B*S, E]
  const int D = in_sizes[0] / M;         // x is [B*S, D]

  float* out = (float*)d_out;                          // router_output [M, 16]
  int*   ids = (int*)((float*)d_out + (size_t)M * EXPERTS);  // ids [M, 2] (int32)

  const int ntiles = (M + 15) / 16;      // one wave per 16-row tile
  const int blocks = (ntiles + 7) / 8;   // 8 waves (256 threads) per block

  noisy_router_kernel<<<blocks, 256, 0, stream>>>(x, noise, Wfc, bfc, Wns, bns,
                                                  out, ids, M, D, ntiles);
}
